// MultiHeadSelfAttention_18296560681487
// MI455X (gfx1250) — matrix operand order repaired and checked by reading, not compiled, not run on hardware
//
#include <hip/hip_runtime.h>
#include <hip/hip_bf16.h>

#define B_  8
#define C_  512
#define T_  1024
#define H_  8
#define CH_ 64

#if __has_builtin(__builtin_amdgcn_sched_barrier)
#define SCHED_FENCE() __builtin_amdgcn_sched_barrier(0)
#else
#define SCHED_FENCE()
#endif

typedef __attribute__((ext_vector_type(16))) _Float16 v16h;
typedef __attribute__((ext_vector_type(8)))  _Float16 v8h;
typedef __attribute__((ext_vector_type(8)))  float    v8f;

union H8   { v8h v; _Float16 e[8]; };
union FAcc { v8f v; float    e[8]; };

__device__ __forceinline__ v8f wmma_f16(v16h a, v16h b, v8f c) {
    return __builtin_amdgcn_wmma_f32_16x16x32_f16(
        false, a, false, b, (short)0, c, false, false);
}

// Packed-fragment LDS layout: tile[sub][kstep][lane(32)][elem(16)] f16.
// A layout (16x32 MxK): element (m, kl) -> lane = m + 16*((kl>>3)&1), e = (kl&7) + 8*(kl>>4)
// B layout (32x16 KxN): element (kl, n) -> lane = n + 16*((kl>>3)&1), e = (kl&7) + 8*(kl>>4)

// ---------------- Kernel 1: GroupNorm -> h (f16) ----------------
__global__ __launch_bounds__(256) void gn_kernel(
        const float* __restrict__ x, const float* __restrict__ gw,
        const float* __restrict__ gb, _Float16* __restrict__ h) {
    const int g = blockIdx.x, b = blockIdx.y, tid = threadIdx.x;
    const size_t base = ((size_t)b * C_ + (size_t)g * 16) * T_;
    const float4* x4 = (const float4*)(x + base);
    float s = 0.f, s2 = 0.f;
    for (int i = tid; i < 16 * T_ / 4; i += 256) {
        float4 xv = x4[i];
        s  += xv.x + xv.y + xv.z + xv.w;
        s2 += xv.x * xv.x + xv.y * xv.y + xv.z * xv.z + xv.w * xv.w;
    }
    __shared__ float r1[256], r2[256];
    r1[tid] = s; r2[tid] = s2;
    __syncthreads();
    for (int off = 128; off > 0; off >>= 1) {
        if (tid < off) { r1[tid] += r1[tid + off]; r2[tid] += r2[tid + off]; }
        __syncthreads();
    }
    const float mu  = r1[0] * (1.f / 16384.f);
    const float var = r2[0] * (1.f / 16384.f) - mu * mu;
    const float rs  = rsqrtf(var + 1e-5f);
    for (int i = tid; i < 16 * T_ / 8; i += 256) {
        int c = g * 16 + (i >> 7);
        float w = gw[c] * rs, bb = gb[c] - mu * w;
        float4 a = x4[i * 2], bv = x4[i * 2 + 1];
        H8 o;
        o.e[0] = (_Float16)(a.x * w + bb);  o.e[1] = (_Float16)(a.y * w + bb);
        o.e[2] = (_Float16)(a.z * w + bb);  o.e[3] = (_Float16)(a.w * w + bb);
        o.e[4] = (_Float16)(bv.x * w + bb); o.e[5] = (_Float16)(bv.y * w + bb);
        o.e[6] = (_Float16)(bv.z * w + bb); o.e[7] = (_Float16)(bv.w * w + bb);
        *(v8h*)(h + base + (size_t)i * 8) = o.v;
    }
}

// ---------------- Kernel 2: QKV GEMM, 128x64 tiles, 2x2 blocked, pipelined ----------------
__global__ __launch_bounds__(256) void qkv_kernel(
        const float* __restrict__ W, const float* __restrict__ bias,
        const _Float16* __restrict__ h,
        _Float16* __restrict__ q, _Float16* __restrict__ k,
        _Float16* __restrict__ v) {
    __shared__ __align__(32) _Float16 Apk[2][8][2][32][16];   // 2 x 16 KB
    __shared__ __align__(32) _Float16 Bpk[2][4][2][32][16];   // 2 x  8 KB
    const int tn = blockIdx.x * 64, tm = blockIdx.y * 128, b = blockIdx.z;
    const int tid = threadIdx.x, wave = tid >> 5, lane = tid & 31;
    const int hi = lane >> 4, mn = lane & 15;
    const int mi0 = (wave >> 1) * 2, ni0 = (wave & 1) * 2;

    float4 aR[8]; v8h bR[2];
    auto globalLoad = [&](int kk) {
        #pragma unroll
        for (int c = 0; c < 4; ++c) {
            int i = tid + 256 * c;
            int r = i >> 3, kc8 = (i & 7) * 8;
            const float4* src = (const float4*)&W[(size_t)(tm + r) * C_ + kk + kc8];
            aR[2 * c] = src[0]; aR[2 * c + 1] = src[1];
        }
        #pragma unroll
        for (int c = 0; c < 2; ++c) {
            int i = tid + 256 * c;
            int kc = i >> 3, n8 = (i & 7) * 8;
            bR[c] = *(const v8h*)&h[((size_t)b * C_ + kk + kc) * T_ + tn + n8];
        }
    };
    auto ldsStore = [&](int buf) {
        #pragma unroll
        for (int c = 0; c < 4; ++c) {
            int i = tid + 256 * c;
            int r = i >> 3, kc8 = (i & 7) * 8;
            float4 x0 = aR[2 * c], x1 = aR[2 * c + 1];
            H8 pk;
            pk.e[0] = (_Float16)x0.x; pk.e[1] = (_Float16)x0.y;
            pk.e[2] = (_Float16)x0.z; pk.e[3] = (_Float16)x0.w;
            pk.e[4] = (_Float16)x1.x; pk.e[5] = (_Float16)x1.y;
            pk.e[6] = (_Float16)x1.z; pk.e[7] = (_Float16)x1.w;
            int kl = kc8 & 31;
            int ahi = (kl >> 3) & 1, e0 = (kl >> 4) << 3;
            *(v8h*)&Apk[buf][r >> 4][kc8 >> 5][(r & 15) + (ahi << 4)][e0] = pk.v;
        }
        #pragma unroll
        for (int c = 0; c < 2; ++c) {
            int i = tid + 256 * c;
            int kc = i >> 3, n8 = (i & 7) * 8;
            H8 pk; pk.v = bR[c];
            int kl = kc & 31;
            int bhi = (kl >> 3) & 1, e = (kl & 7) + ((kl >> 4) << 3);
            _Float16 (*dst)[16] = Bpk[buf][n8 >> 4][kc >> 5];
            #pragma unroll
            for (int j = 0; j < 8; ++j) dst[((n8 & 15) + j) + (bhi << 4)][e] = pk.e[j];
        }
    };

    v8f a00 = {}, a01 = {}, a10 = {}, a11 = {};
    globalLoad(0);
    ldsStore(0);
    __syncthreads();
    for (int s = 0; s < C_ / 64; ++s) {
        const int buf = s & 1;
        if (s < C_ / 64 - 1) globalLoad((s + 1) * 64);
        SCHED_FENCE();   // keep next-stage global loads issued before the WMMA block
        #pragma unroll
        for (int ks = 0; ks < 2; ++ks) {
            v16h fa0 = *(const v16h*)&Apk[buf][mi0][ks][lane][0];
            v16h fa1 = *(const v16h*)&Apk[buf][mi0 + 1][ks][lane][0];
            v16h fb0 = *(const v16h*)&Bpk[buf][ni0][ks][lane][0];
            v16h fb1 = *(const v16h*)&Bpk[buf][ni0 + 1][ks][lane][0];
            a00 = wmma_f16(fa0, fb0, a00);
            a01 = wmma_f16(fa0, fb1, a01);
            a10 = wmma_f16(fa1, fb0, a10);
            a11 = wmma_f16(fa1, fb1, a11);
        }
        SCHED_FENCE();   // keep convert/store phase after the WMMA block
        if (s < C_ / 64 - 1) ldsStore(buf ^ 1);
        __syncthreads();
    }
    const float scale = 0.35355339059327373f; // 1/ch^0.25
    #pragma unroll
    for (int p = 0; p < 4; ++p) {
        const int mi = mi0 + (p >> 1), ni = ni0 + (p & 1);
        FAcc acc; acc.v = (p == 0) ? a00 : (p == 1) ? a01 : (p == 2) ? a10 : a11;
        #pragma unroll
        for (int r = 0; r < 8; ++r) {
            int o = tm + mi * 16 + (hi ? 8 : 0) + r;
            int t = tn + ni * 16 + mn;
            float val = acc.e[r] + bias[o];
            int head = o / 192, rr = o % 192;
            size_t bh = (size_t)b * H_ + head;
            if (rr < 64)
                q[(bh * CH_ + rr) * T_ + t] = (_Float16)(val * scale);
            else if (rr < 128)
                k[(bh * CH_ + (rr - 64)) * T_ + t] = (_Float16)(val * scale);
            else
                v[(bh * CH_ + (rr - 128)) * T_ + t] = (_Float16)val;
        }
    }
}

// ---------------- Kernel 3: flash attention per (bh, 64-query tile) ----------------
__global__ __launch_bounds__(256) void attn_kernel(
        const _Float16* __restrict__ q, const _Float16* __restrict__ k,
        const _Float16* __restrict__ v, _Float16* __restrict__ aout) {
    __shared__ __align__(32) _Float16 QtP[4][2][32][16];   // A of S = Q^T K
    __shared__ __align__(32) _Float16 KsP[4][2][32][16];   // B of S
    __shared__ __align__(32) _Float16 VtP[4][2][32][16];   // B of O = P V^T
    __shared__ __align__(32) _Float16 PbP[4][2][32][16];   // A of O
    __shared__ float Sb[64][65];
    __shared__ float rmax[64], rsum[64], rfac[64];
    const int t0 = blockIdx.x * 64;
    const int bh = blockIdx.y;
    const int tid = threadIdx.x, wave = tid >> 5, lane = tid & 31;
    const int hi = lane >> 4, mn = lane & 15;
    const int sub0 = wave * 2, sub1 = sub0 + 1;
    const int ma = sub0 >> 2, na = sub0 & 3, mb = sub1 >> 2, nb = sub1 & 3;

    for (int i = tid; i < 512; i += 256) {
        int c = i >> 3, t8 = (i & 7) * 8;
        H8 pk; pk.v = *(const v8h*)&q[((size_t)bh * CH_ + c) * T_ + t0 + t8];
        int kl = c & 31;
        int ahi = (kl >> 3) & 1, e = (kl & 7) + ((kl >> 4) << 3);
        _Float16 (*dst)[16] = QtP[t8 >> 4][c >> 5];
        #pragma unroll
        for (int j = 0; j < 8; ++j) dst[((t8 & 15) + j) + (ahi << 4)][e] = pk.e[j];
    }
    if (tid < 64) { rmax[tid] = -1e30f; rsum[tid] = 0.f; }
    v8f o0 = {}; v8f o1 = {};

    for (int st = 0; st < 16; ++st) {
        const int s0 = st * 64;
        __syncthreads();
        for (int i = tid; i < 512; i += 256) {
            int c = i >> 3, s8 = (i & 7) * 8;
            H8 pk; pk.v = *(const v8h*)&k[((size_t)bh * CH_ + c) * T_ + s0 + s8];
            int kl = c & 31;
            int bhi = (kl >> 3) & 1, e = (kl & 7) + ((kl >> 4) << 3);
            _Float16 (*dst)[16] = KsP[s8 >> 4][c >> 5];
            #pragma unroll
            for (int j = 0; j < 8; ++j) dst[((s8 & 15) + j) + (bhi << 4)][e] = pk.e[j];
        }
        for (int i = tid; i < 512; i += 256) {
            int c = i >> 3, s8 = (i & 7) * 8;
            H8 pk; pk.v = *(const v8h*)&v[((size_t)bh * CH_ + c) * T_ + s0 + s8];
            int kl = s8 & 31;
            int bhi = (kl >> 3) & 1, e0 = (kl >> 4) << 3;
            *(v8h*)&VtP[c >> 4][s8 >> 5][(c & 15) + (bhi << 4)][e0] = pk.v;
        }
        __syncthreads();
        #pragma unroll
        for (int s = 0; s < 2; ++s) {
            const int mi = (s == 0) ? ma : mb, ni = (s == 0) ? na : nb;
            v8f sc = {};
            #pragma unroll
            for (int ks = 0; ks < 2; ++ks)
                sc = wmma_f16(*(const v16h*)&QtP[mi][ks][lane][0],
                              *(const v16h*)&KsP[ni][ks][lane][0], sc);
            FAcc sa; sa.v = sc;
            #pragma unroll
            for (int r = 0; r < 8; ++r)
                Sb[mi * 16 + (hi ? 8 : 0) + r][ni * 16 + mn] = sa.e[r];
        }
        __syncthreads();
        {
            const int row = tid >> 2, qq = tid & 3;
            const float mo = rmax[row];
            float vloc[16], mx = mo;
            #pragma unroll
            for (int j = 0; j < 16; ++j) {
                vloc[j] = Sb[row][qq * 16 + j];
                mx = fmaxf(mx, vloc[j]);
            }
            mx = fmaxf(mx, __shfl_xor(mx, 1, 4));
            mx = fmaxf(mx, __shfl_xor(mx, 2, 4));
            const float f = __expf(mo - mx);
            H8 pa, pb;
            float sum = 0.f;
            #pragma unroll
            for (int j = 0; j < 8; ++j) {
                float p = __expf(vloc[j] - mx); sum += p; pa.e[j] = (_Float16)p;
            }
            #pragma unroll
            for (int j = 0; j < 8; ++j) {
                float p = __expf(vloc[8 + j] - mx); sum += p; pb.e[j] = (_Float16)p;
            }
            sum += __shfl_xor(sum, 1, 4);
            sum += __shfl_xor(sum, 2, 4);
            const int msub = row >> 4, m = row & 15, ksp = qq >> 1, e0 = (qq & 1) * 8;
            *(v8h*)&PbP[msub][ksp][m][e0]      = pa.v;
            *(v8h*)&PbP[msub][ksp][m + 16][e0] = pb.v;
            if (qq == 0) { rsum[row] = rsum[row] * f + sum; rmax[row] = mx; rfac[row] = f; }
        }
        __syncthreads();
        #pragma unroll
        for (int s = 0; s < 2; ++s) {
            const int mi = (s == 0) ? ma : mb, ni = (s == 0) ? na : nb;
            v8f oc = (s == 0) ? o0 : o1;
            FAcc oa; oa.v = oc;
            #pragma unroll
            for (int r = 0; r < 8; ++r) oa.e[r] *= rfac[mi * 16 + (hi ? 8 : 0) + r];
            oc = oa.v;
            #pragma unroll
            for (int ks = 0; ks < 2; ++ks)
                oc = wmma_f16(*(const v16h*)&PbP[mi][ks][lane][0],
                              *(const v16h*)&VtP[ni][ks][lane][0], oc);
            if (s == 0) o0 = oc; else o1 = oc;
        }
    }
    __syncthreads();
    const int b = bh >> 3, head = bh & 7;
    #pragma unroll
    for (int s = 0; s < 2; ++s) {
        const int mi = (s == 0) ? ma : mb, ni = (s == 0) ? na : nb;
        FAcc oa; oa.v = (s == 0) ? o0 : o1;
        #pragma unroll
        for (int r = 0; r < 8; ++r) {
            int tl = mi * 16 + (hi ? 8 : 0) + r;
            int cl = ni * 16 + mn;
            float val = oa.e[r] / rsum[tl];
            aout[((size_t)b * C_ + head * CH_ + cl) * T_ + t0 + tl] = (_Float16)val;
        }
    }
}

// ---------------- Kernel 4: proj GEMM + residual, 128x64 tiles, pipelined ----------------
__global__ __launch_bounds__(256) void proj_kernel(
        const float* __restrict__ W, const float* __restrict__ bias,
        const _Float16* __restrict__ a, const float* __restrict__ x,
        float* __restrict__ out) {
    __shared__ __align__(32) _Float16 Apk[2][8][2][32][16];
    __shared__ __align__(32) _Float16 Bpk[2][4][2][32][16];
    const int tn = blockIdx.x * 64, tm = blockIdx.y * 128, b = blockIdx.z;
    const int tid = threadIdx.x, wave = tid >> 5, lane = tid & 31;
    const int hi = lane >> 4, mn = lane & 15;
    const int mi0 = (wave >> 1) * 2, ni0 = (wave & 1) * 2;

    float4 aR[8]; v8h bR[2];
    auto globalLoad = [&](int kk) {
        #pragma unroll
        for (int c = 0; c < 4; ++c) {
            int i = tid + 256 * c;
            int r = i >> 3, kc8 = (i & 7) * 8;
            const float4* src = (const float4*)&W[(size_t)(tm + r) * C_ + kk + kc8];
            aR[2 * c] = src[0]; aR[2 * c + 1] = src[1];
        }
        #pragma unroll
        for (int c = 0; c < 2; ++c) {
            int i = tid + 256 * c;
            int kc = i >> 3, n8 = (i & 7) * 8;
            bR[c] = *(const v8h*)&a[((size_t)b * C_ + kk + kc) * T_ + tn + n8];
        }
    };
    auto ldsStore = [&](int buf) {
        #pragma unroll
        for (int c = 0; c < 4; ++c) {
            int i = tid + 256 * c;
            int r = i >> 3, kc8 = (i & 7) * 8;
            float4 x0 = aR[2 * c], x1 = aR[2 * c + 1];
            H8 pk;
            pk.e[0] = (_Float16)x0.x; pk.e[1] = (_Float16)x0.y;
            pk.e[2] = (_Float16)x0.z; pk.e[3] = (_Float16)x0.w;
            pk.e[4] = (_Float16)x1.x; pk.e[5] = (_Float16)x1.y;
            pk.e[6] = (_Float16)x1.z; pk.e[7] = (_Float16)x1.w;
            int kl = kc8 & 31;
            int ahi = (kl >> 3) & 1, e0 = (kl >> 4) << 3;
            *(v8h*)&Apk[buf][r >> 4][kc8 >> 5][(r & 15) + (ahi << 4)][e0] = pk.v;
        }
        #pragma unroll
        for (int c = 0; c < 2; ++c) {
            int i = tid + 256 * c;
            int kc = i >> 3, n8 = (i & 7) * 8;
            H8 pk; pk.v = bR[c];
            int kl = kc & 31;
            int bhi = (kl >> 3) & 1, e = (kl & 7) + ((kl >> 4) << 3);
            _Float16 (*dst)[16] = Bpk[buf][n8 >> 4][kc >> 5];
            #pragma unroll
            for (int j = 0; j < 8; ++j) dst[((n8 & 15) + j) + (bhi << 4)][e] = pk.e[j];
        }
    };

    v8f a00 = {}, a01 = {}, a10 = {}, a11 = {};
    globalLoad(0);
    ldsStore(0);
    __syncthreads();
    for (int s = 0; s < C_ / 64; ++s) {
        const int buf = s & 1;
        if (s < C_ / 64 - 1) globalLoad((s + 1) * 64);
        SCHED_FENCE();
        #pragma unroll
        for (int ks = 0; ks < 2; ++ks) {
            v16h fa0 = *(const v16h*)&Apk[buf][mi0][ks][lane][0];
            v16h fa1 = *(const v16h*)&Apk[buf][mi0 + 1][ks][lane][0];
            v16h fb0 = *(const v16h*)&Bpk[buf][ni0][ks][lane][0];
            v16h fb1 = *(const v16h*)&Bpk[buf][ni0 + 1][ks][lane][0];
            a00 = wmma_f16(fa0, fb0, a00);
            a01 = wmma_f16(fa0, fb1, a01);
            a10 = wmma_f16(fa1, fb0, a10);
            a11 = wmma_f16(fa1, fb1, a11);
        }
        SCHED_FENCE();
        if (s < C_ / 64 - 1) ldsStore(buf ^ 1);
        __syncthreads();
    }
    #pragma unroll
    for (int p = 0; p < 4; ++p) {
        const int mi = mi0 + (p >> 1), ni = ni0 + (p & 1);
        FAcc acc; acc.v = (p == 0) ? a00 : (p == 1) ? a01 : (p == 2) ? a10 : a11;
        #pragma unroll
        for (int r = 0; r < 8; ++r) {
            int o = tm + mi * 16 + (hi ? 8 : 0) + r;
            int t = tn + ni * 16 + mn;
            size_t idx = ((size_t)b * C_ + o) * T_ + t;
            out[idx] = x[idx] + acc.e[r] + bias[o];
        }
    }
}

extern "C" void kernel_launch(void* const* d_in, const int* in_sizes, int n_in,
                              void* d_out, int out_size, void* d_ws, size_t ws_size,
                              hipStream_t stream) {
    (void)in_sizes; (void)n_in; (void)out_size; (void)ws_size;
    const float* x      = (const float*)d_in[0];
    const float* gn_w   = (const float*)d_in[1];
    const float* gn_b   = (const float*)d_in[2];
    const float* qkv_w  = (const float*)d_in[3];
    const float* qkv_b  = (const float*)d_in[4];
    const float* proj_w = (const float*)d_in[5];
    const float* proj_b = (const float*)d_in[6];
    float* out = (float*)d_out;

    const size_t NBCT = (size_t)B_ * C_ * T_;
    _Float16* h  = (_Float16*)d_ws;
    _Float16* qf = h  + NBCT;
    _Float16* kf = qf + NBCT;
    _Float16* vf = kf + NBCT;
    _Float16* af = vf + NBCT;   // 40 MB workspace

    gn_kernel<<<dim3(32, B_), 256, 0, stream>>>(x, gn_w, gn_b, h);
    qkv_kernel<<<dim3(T_ / 64, (3 * C_) / 128, B_), 256, 0, stream>>>(
        qkv_w, qkv_b, h, qf, kf, vf);
    attn_kernel<<<dim3(T_ / 64, B_ * H_), 256, 0, stream>>>(qf, kf, vf, af);
    proj_kernel<<<dim3(T_ / 64, C_ / 128, B_), 256, 0, stream>>>(
        proj_w, proj_b, af, x, out);
}
